// MultiHeadAttention_83958020702759
// MI455X (gfx1250) — matrix operand order repaired and checked by reading, not compiled, not run on hardware
//
#include <hip/hip_runtime.h>

// ---------------------------------------------------------------------------
// MI455X (gfx1250) multi-head attention, bf16 WMMA + double-buffered TDM.
// S=1024, B=4, D=1024, H=16, DH=64.  Activation rows r = s*B+b  [4096 x 1024].
//
// Pipeline:
//   1) one-time converts: activations f32->bf16; weights f32->bf16 transposed
//      to B^T [N=1024][K=1024] (head-blocked Wq/Wk/Wv remapped in the same pass)
//   2) three WMMA GEMMs (bf16 in, bf16 out) for Q/K/V projections
//   3) flash-attention (one wave per 16 q-rows x b x h), online softmax,
//      score + P@V WMMAs, bf16 concat out
//   4) WMMA GEMM (bf16 in, f32 out + bias) for the output projection
// GEMM tiles staged by the Tensor Data Mover (tensor_load_to_lds), double
// buffered so the DMA of tile i+1 overlaps WMMA compute on tile i.  The
// LDS tile base is laundered through an asm so the TDM's writes (visible
// only via scalar descriptors) cannot be "proven absent" by alias analysis,
// and reloaded through an addrspace(3) pointer so accesses stay ds_loads.
// ---------------------------------------------------------------------------

typedef __attribute__((ext_vector_type(16))) __bf16 v16bf;
typedef __attribute__((ext_vector_type(8)))  float  v8f;
typedef unsigned u32x4 __attribute__((ext_vector_type(4)));
typedef int      i32x4 __attribute__((ext_vector_type(4)));
typedef int      i32x8 __attribute__((ext_vector_type(8)));

typedef const __attribute__((address_space(3))) unsigned lds_cu32;

union BF16Frag { v16bf v; unsigned u[8]; };

#if __has_builtin(__builtin_amdgcn_tensor_load_to_lds) && \
    __has_builtin(__builtin_amdgcn_s_wait_tensorcnt)
  #define HAVE_TDM 1
  #if __has_include(<hip/amd_detail/amd_gfx1250_TDM.h>)
    #define TDM_ARITY6 1   // amdgpu-toolchain (therock headers): 6-arg builtin
  #endif
#endif

__device__ __forceinline__ unsigned pack2_bf16(float a, float b) {
  union { float f; unsigned u; } xa, xb;
  xa.f = a; xb.f = b;
  unsigned ha = (xa.u + 0x7FFFu + ((xa.u >> 16) & 1u)) >> 16;
  unsigned hb = (xb.u + 0x7FFFu + ((xb.u >> 16) & 1u)) >> 16;
  return (ha & 0xFFFFu) | (hb << 16);
}

__device__ __forceinline__ unsigned short f_to_bf16(float a) {
  union { float f; unsigned u; } x; x.f = a;
  return (unsigned short)((x.u + 0x7FFFu + ((x.u >> 16) & 1u)) >> 16);
}

__device__ __forceinline__ v8f vzero8() {
  v8f z = {0.f, 0.f, 0.f, 0.f, 0.f, 0.f, 0.f, 0.f};
  return z;
}

// ---------------------------------------------------------------------------
// Elementwise f32 -> bf16 (4M elements)
// ---------------------------------------------------------------------------
__global__ __launch_bounds__(256)
void convert_bf16(const float* __restrict__ X, unsigned short* __restrict__ Y,
                  int n) {
  const int i = (blockIdx.x * 256 + threadIdx.x) * 8;
  if (i + 8 <= n) {
#pragma unroll
    for (int j = 0; j < 8; j += 2)
      *(unsigned*)&Y[i + j] = pack2_bf16(X[i + j], X[i + j + 1]);
  }
}

// ---------------------------------------------------------------------------
// Weight convert+transpose: WT[c][d] = W(d, c) as bf16.
// mode 0: W is [H, D, DH] head-blocked -> W(d, c) = W[(c>>6)*65536 + d*64 + (c&63)]
// mode 1: W is [1024, 1024] row-major  -> W(d, c) = W[d*1024 + c]
// ---------------------------------------------------------------------------
__global__ __launch_bounds__(256)
void convert_transpose_w(const float* __restrict__ W,
                         unsigned short* __restrict__ WT, int mode) {
  __shared__ float tile[32][33];
  const int tx = threadIdx.x;            // 0..31
  const int ty = threadIdx.y;            // 0..7
  const int c0 = blockIdx.x * 32;
  const int d0 = blockIdx.y * 32;
#pragma unroll
  for (int i = ty; i < 32; i += 8) {
    const int d = d0 + i;
    const int c = c0 + tx;
    float v;
    if (mode == 0)
      v = W[(size_t)(c >> 6) * 65536 + (size_t)d * 64 + (c & 63)];
    else
      v = W[(size_t)d * 1024 + c];
    tile[i][tx] = v;
  }
  __syncthreads();
#pragma unroll
  for (int i = ty; i < 32; i += 8)
    WT[(size_t)(c0 + i) * 1024 + d0 + tx] = f_to_bf16(tile[tx][i]);
}

// ---------------------------------------------------------------------------
// WMMA GEMM: C[4096x1024] = A[4096x1024] * B + bias, A and B^T bf16 row-major.
// BT is [col][K] so both tiles stage as identical contiguous 128x32 copies.
// Block = 128x128, 256 thr = 8 waves, wave (wm,wn) owns 32x64 (8 accumulators).
// LDS double buffer: half h -> A tile @ base+h*16384, B tile @ +8192,
// each [row128][kpair16] uints.  Wave 0 drives the TDM.
// ---------------------------------------------------------------------------
__global__ __launch_bounds__(256)
void mha_gemm_wmma(const unsigned short* __restrict__ A,
                   const unsigned short* __restrict__ BT,
                   const float* __restrict__ bias,
                   void* __restrict__ Cout, int out_bf16) {
  __shared__ unsigned smem32[8192];           // 32 KB: two (A 8K | B 8K) halves

  const int tid  = threadIdx.x;
  const int lane = tid & 31;
  const int w    = tid >> 5;
  const int wm   = w & 3;
  const int wn   = w >> 2;
  const int lo   = lane & 15;
  const int hi   = lane >> 4;
  const int blockM = blockIdx.y * 128;
  const int blockN = blockIdx.x * 128;
  const bool isw0  = (tid < 32);

  // Workgroup-relative LDS byte address of smem32 (low 32 bits of the generic
  // pointer per the ISA's flat->LDS aperture mapping).  readfirstlane gives a
  // legal VGPR->SGPR move so the TDM descriptors can build in SALU.
  const unsigned ldsbase = (unsigned)__builtin_amdgcn_readfirstlane(
      (int)(unsigned)(unsigned long long)(const void*)smem32);

  v8f acc[2][4];
#pragma unroll
  for (int m = 0; m < 2; ++m)
#pragma unroll
    for (int n = 0; n < 4; ++n) acc[m][n] = vzero8();

#if defined(HAVE_TDM)
  // D# group1 (loop-invariant): 2D, data_size=2B, tile 32x128, stride 1024.
  i32x8 g1;
  g1[0] = 0x00010000;            // data_size=1 (2 bytes)
  g1[1] = (int)(1024u << 16);    // tensor_dim0 = 1024 (low 16 bits)
  g1[2] = (int)(4096u << 16);    // tensor_dim0 hi=0 | tensor_dim1 = 4096
  g1[3] = (int)(32u << 16);      // tile_dim0 = 32
  g1[4] = 128;                   // tile_dim1 = 128, tile_dim2 = 0
  g1[5] = 1024;                  // tensor_dim0_stride = 1024 elements
  g1[6] = 0;
  g1[7] = 0;
  const i32x4 gz = {0, 0, 0, 0};
  #if defined(TDM_ARITY6)
  const i32x8 gz8 = {0, 0, 0, 0, 0, 0, 0, 0};
  #endif
  const unsigned long long gA0 =
      (unsigned long long)(const void*)(A + ((size_t)blockM << 10));
  const unsigned long long gB0 =
      (unsigned long long)(const void*)(BT + ((size_t)blockN << 10));

  auto tdm_issue = [&](int it, int half) {
    const unsigned long long off = (unsigned long long)(it * 64);  // 32 elem * 2B
    const unsigned long long ga = gA0 + off;
    const unsigned long long gb = gB0 + off;
    u32x4 g0;
    g0[0] = 1u;                                   // count=1 (valid user D#)
    g0[1] = ldsbase + (unsigned)(half * 16384);   // lds_addr: A tile
    g0[2] = (unsigned)ga;
    g0[3] = (unsigned)(ga >> 32) | 0x80000000u;   // addr[56:32] | type=2
  #if defined(TDM_ARITY6)
    __builtin_amdgcn_tensor_load_to_lds(g0, g1, gz, gz, gz8, 0);
  #else
    __builtin_amdgcn_tensor_load_to_lds(g0, g1, gz, gz, 0);
  #endif
    g0[1] = ldsbase + (unsigned)(half * 16384 + 8192);   // lds_addr: B tile
    g0[2] = (unsigned)gb;
    g0[3] = (unsigned)(gb >> 32) | 0x80000000u;
  #if defined(TDM_ARITY6)
    __builtin_amdgcn_tensor_load_to_lds(g0, g1, gz, gz, gz8, 0);
  #else
    __builtin_amdgcn_tensor_load_to_lds(g0, g1, gz, gz, 0);
  #endif
  };

  if (isw0) tdm_issue(0, 0);   // prologue: DMA tile 0 into half 0
#endif

#pragma unroll 1
  for (int it = 0; it < 32; ++it) {
    const int half = it & 1;
#if defined(HAVE_TDM)
    if (isw0) __builtin_amdgcn_s_wait_tensorcnt(0);   // tile `it` landed
    __syncthreads();                                   // publish to all waves
    if (isw0 && (it + 1) < 32)
      tdm_issue(it + 1, half ^ 1);    // overlap next DMA with this compute
    // Launder the tile base so the TDM's LDS writes cannot be reasoned away
    // (the shared array has no visible stores); reload via addrspace(3) so
    // the accesses still lower to ds_load.  "+v": VGPR, legal copy direction.
    unsigned abase = ldsbase + (unsigned)(half * 16384);
    asm volatile("" : "+v"(abase) : : "memory");
    lds_cu32* As32 = (lds_cu32*)abase;
    lds_cu32* Bs32 = (lds_cu32*)(abase + 8192u);
#else
    __syncthreads();   // all waves done reading previous tile
    {
      const unsigned short* Asrc = A + ((size_t)blockM << 10) + it * 32;
      const unsigned short* Bsrc = BT + ((size_t)blockN << 10) + it * 32;
#pragma unroll
      for (int ss = tid; ss < 512; ss += 256) {
        const int row = ss >> 2, part = ss & 3;
        *(uint4*)((char*)smem32 + row * 64 + part * 16) =
            *(const uint4*)((const char*)(Asrc + ((size_t)row << 10)) + part * 16);
        *(uint4*)((char*)smem32 + 8192 + row * 64 + part * 16) =
            *(const uint4*)((const char*)(Bsrc + ((size_t)row << 10)) + part * 16);
      }
    }
    __syncthreads();
    const unsigned* As32 = smem32;
    const unsigned* Bs32 = smem32 + 2048;
#endif

    // A fragments: lane -> row M = lane&15; K = 16*hi + 2j
    BF16Frag af[2];
#pragma unroll
    for (int m = 0; m < 2; ++m) {
      const int row = wm * 32 + m * 16 + lo;
#pragma unroll
      for (int j = 0; j < 4; ++j) {
        af[m].u[j]     = As32[row * 16 + 8 * hi + j];
        af[m].u[j + 4] = As32[row * 16 + 8 * hi + 4 + j];
      }
    }
    // B fragments: lane -> col N = lane&15; K = 16*hi + 2j
#pragma unroll
    for (int n = 0; n < 4; ++n) {
      BF16Frag bf;
      const int col = wn * 64 + n * 16 + lo;
#pragma unroll
      for (int j = 0; j < 8; ++j)
        bf.u[j] = Bs32[col * 16 + 8 * hi + j];
#pragma unroll
      for (int m = 0; m < 2; ++m)
        acc[m][n] = __builtin_amdgcn_wmma_f32_16x16x32_bf16(
            false, af[m].v, false, bf.v, (short)0, acc[m][n], false, false);
    }
  }

  // epilogue: C layout VGPR r -> row 8*hi + r, col lane&15
  float* Cf = (float*)Cout;
  unsigned short* Ch = (unsigned short*)Cout;
#pragma unroll
  for (int m = 0; m < 2; ++m) {
#pragma unroll
    for (int n = 0; n < 4; ++n) {
      const int col = blockN + wn * 64 + n * 16 + lo;
      const float bb = bias[col];
#pragma unroll
      for (int r = 0; r < 8; ++r) {
        const int row = blockM + wm * 32 + m * 16 + hi * 8 + r;
        const float vv = acc[m][n][r] + bb;
        if (out_bf16) Ch[(size_t)row * 1024 + col] = f_to_bf16(vv);
        else          Cf[(size_t)row * 1024 + col] = vv;
      }
    }
  }
}

// ---------------------------------------------------------------------------
// Flash attention, one wave per (16 q-rows, b, h). Q/K/V are bf16 [4096][1024].
// 32-key chunks: 4 score WMMAs, online softmax (shfl_xor butterflies inside
// the 16-lane row groups), P restaged via LDS into A layout, 4 P@V WMMAs.
// ---------------------------------------------------------------------------
__global__ __launch_bounds__(32)
void mha_flash_attn(const unsigned short* __restrict__ Q,
                    const unsigned short* __restrict__ K,
                    const unsigned short* __restrict__ V,
                    const int* __restrict__ mask,
                    unsigned short* __restrict__ O) {
  __shared__ unsigned short Plds[16 * 32];

  const int lane = threadIdx.x;
  const int lo = lane & 15;
  const int hi = lane >> 4;
  const int s0 = blockIdx.x * 16;
  const int b  = blockIdx.y;
  const int h  = blockIdx.z;
  const int hd = h * 64;
  const float scale = 0.125f;   // 1/sqrt(64)

  // Q as two A fragments; pairs are adjacent bf16 -> single dword loads
  BF16Frag qf[2];
  {
    const unsigned short* qrow = Q + ((size_t)(s0 + lo) * 4 + b) * 1024 + hd;
#pragma unroll
    for (int t = 0; t < 2; ++t)
#pragma unroll
      for (int j = 0; j < 8; ++j) {
        const int kidx = 16 * hi + 2 * j;
        qf[t].u[j] = *(const unsigned*)&qrow[t * 32 + kidx];
      }
  }

  v8f acc[4];
#pragma unroll
  for (int n = 0; n < 4; ++n) acc[n] = vzero8();
  float m_i[8], l_i[8];
#pragma unroll
  for (int r = 0; r < 8; ++r) { m_i[r] = -3.0e38f; l_i[r] = 0.0f; }

  for (int j0 = 0; j0 < 1024; j0 += 32) {
    // scores: two 16x16 tiles
    v8f sc[2];
#pragma unroll
    for (int t = 0; t < 2; ++t) {
      v8f c = vzero8();
#pragma unroll
      for (int kh = 0; kh < 2; ++kh) {
        BF16Frag bf;   // B = K^T: N = key (lane&15), K-dim = dh
        const unsigned short* krow =
            K + ((size_t)(j0 + t * 16 + lo) * 4 + b) * 1024 + hd + kh * 32;
#pragma unroll
        for (int j = 0; j < 8; ++j)
          bf.u[j] = *(const unsigned*)&krow[16 * hi + 2 * j];
        c = __builtin_amdgcn_wmma_f32_16x16x32_bf16(
            false, qf[kh].v, false, bf.v, (short)0, c, false, false);
      }
      sc[t] = c;
    }

    const float mk0 = (float)mask[(size_t)(j0 + lo) * 4 + b] * 1e18f;
    const float mk1 = (float)mask[(size_t)(j0 + 16 + lo) * 4 + b] * 1e18f;

    float p0[8], p1[8], alpha[8];
#pragma unroll
    for (int r = 0; r < 8; ++r) {           // row M = 8*hi + r
      float a0 = sc[0][r] * scale - mk0;
      float a1 = sc[1][r] * scale - mk1;
      float mx = fmaxf(a0, a1);
#pragma unroll
      for (int d = 1; d <= 8; d <<= 1)
        mx = fmaxf(mx, __shfl_xor(mx, d, 32));
      const float mnew = fmaxf(m_i[r], mx);
      const float al = __expf(m_i[r] - mnew);
      const float e0 = __expf(a0 - mnew);
      const float e1 = __expf(a1 - mnew);
      float rs = e0 + e1;
#pragma unroll
      for (int d = 1; d <= 8; d <<= 1)
        rs += __shfl_xor(rs, d, 32);
      l_i[r] = l_i[r] * al + rs;
      m_i[r] = mnew;
      alpha[r] = al;
      p0[r] = e0; p1[r] = e1;
    }
#pragma unroll
    for (int n = 0; n < 4; ++n)
#pragma unroll
      for (int r = 0; r < 8; ++r)
        acc[n][r] *= alpha[r];

    // restage P (C layout) into LDS, regather as A fragment
    __syncthreads();
#pragma unroll
    for (int r = 0; r < 8; ++r) {
      Plds[(hi * 8 + r) * 32 + lo]      = f_to_bf16(p0[r]);
      Plds[(hi * 8 + r) * 32 + 16 + lo] = f_to_bf16(p1[r]);
    }
    __syncthreads();

    BF16Frag pf;
#pragma unroll
    for (int j = 0; j < 8; ++j) {
      const int kidx = 16 * hi + 2 * j;
      pf.u[j] = (unsigned)Plds[lo * 32 + kidx] |
                ((unsigned)Plds[lo * 32 + kidx + 1] << 16);
    }

    // acc += P[16x32] @ V[32x64]
#pragma unroll
    for (int n = 0; n < 4; ++n) {
      BF16Frag vf;   // B frag: N = dh col, K = key in chunk (rows 4096 apart)
#pragma unroll
      for (int j = 0; j < 8; ++j) {
        const unsigned short* vp =
            V + ((size_t)(j0 + 16 * hi + 2 * j) * 4 + b) * 1024 + hd + n * 16 + lo;
        vf.u[j] = (unsigned)vp[0] | ((unsigned)vp[4096] << 16);
      }
      acc[n] = __builtin_amdgcn_wmma_f32_16x16x32_bf16(
          false, pf.v, false, vf.v, (short)0, acc[n], false, false);
    }
  }

  // normalize + write bf16 concat tile
#pragma unroll
  for (int r = 0; r < 8; ++r) {
    const float inv = 1.0f / l_i[r];
    const size_t row = (size_t)(s0 + hi * 8 + r) * 4 + b;
#pragma unroll
    for (int n = 0; n < 4; ++n)
      O[row * 1024 + hd + n * 16 + lo] = f_to_bf16(acc[n][r] * inv);
  }
}

// ---------------------------------------------------------------------------
extern "C" void kernel_launch(void* const* d_in, const int* in_sizes, int n_in,
                              void* d_out, int out_size, void* d_ws, size_t ws_size,
                              hipStream_t stream) {
  (void)in_sizes; (void)n_in; (void)out_size; (void)ws_size;

  const float* query = (const float*)d_in[0];
  const float* key_  = (const float*)d_in[1];
  const float* value = (const float*)d_in[2];
  const int*   kmask = (const int*)d_in[3];
  const float* Wq = (const float*)d_in[4];
  const float* bq = (const float*)d_in[5];
  const float* Wk = (const float*)d_in[6];
  const float* bk = (const float*)d_in[7];
  const float* Wv = (const float*)d_in[8];
  const float* bv = (const float*)d_in[9];
  const float* Wo = (const float*)d_in[10];
  const float* bo = (const float*)d_in[11];
  float* out = (float*)d_out;

  unsigned short* us = (unsigned short*)d_ws;
  const size_t plane = (size_t)4096 * 1024;
  const size_t wpl   = (size_t)1024 * 1024;
  unsigned short* xq  = us;               // bf16 activations
  unsigned short* xk  = xq  + plane;
  unsigned short* xv  = xk  + plane;
  unsigned short* WqT = xv  + plane;      // bf16 transposed weights [N][K]
  unsigned short* WkT = WqT + wpl;
  unsigned short* WvT = WkT + wpl;
  unsigned short* WoT = WvT + wpl;
  unsigned short* qp  = WoT + wpl;        // projected q/k/v (bf16)
  unsigned short* kp  = qp  + plane;
  unsigned short* vp  = kp  + plane;
  unsigned short* cc  = vp  + plane;      // concat (bf16)

  const int NACT = 4096 * 1024;
  convert_bf16<<<2048, 256, 0, stream>>>(query, xq, NACT);
  convert_bf16<<<2048, 256, 0, stream>>>(key_,  xk, NACT);
  convert_bf16<<<2048, 256, 0, stream>>>(value, xv, NACT);

  dim3 gT(32, 32), bT(32, 8);
  convert_transpose_w<<<gT, bT, 0, stream>>>(Wq, WqT, 0);
  convert_transpose_w<<<gT, bT, 0, stream>>>(Wk, WkT, 0);
  convert_transpose_w<<<gT, bT, 0, stream>>>(Wv, WvT, 0);
  convert_transpose_w<<<gT, bT, 0, stream>>>(Wo, WoT, 1);

  dim3 gGemm(8, 32), bGemm(256);
  mha_gemm_wmma<<<gGemm, bGemm, 0, stream>>>(xq, WqT, bq, qp, 1);
  mha_gemm_wmma<<<gGemm, bGemm, 0, stream>>>(xk, WkT, bk, kp, 1);
  mha_gemm_wmma<<<gGemm, bGemm, 0, stream>>>(xv, WvT, bv, vp, 1);

  mha_flash_attn<<<dim3(64, 4, 16), 32, 0, stream>>>(qp, kp, vp, kmask, cc);

  mha_gemm_wmma<<<gGemm, bGemm, 0, stream>>>(cc, WoT, bo, out, 0);
}
